// SSD_7713761264355
// MI455X (gfx1250) — matrix operand order repaired and checked by reading, not compiled, not run on hardware
//
#include <hip/hip_runtime.h>
#include <hip/hip_bf16.h>

// ---------------- problem constants ----------------
#define B_SZ   2
#define L_SZ   4096
#define DM     1024
#define DST    128
#define KCONV  4
#define HD     64
#define DIN    2048
#define DSSM   2048
#define NH     32
#define CONVD  2304           // DSSM + 2*DST
#define DINP   4384           // 2*DIN + 2*DST + NH
#define CHUNK  256
#define NC     16             // L / CHUNK
#define BL     (B_SZ * L_SZ)  // 8192 rows
#define BC_N   (B_SZ * NC)    // 32 (b,chunk) pairs

typedef __attribute__((ext_vector_type(16))) __bf16 v16bf;
typedef __attribute__((ext_vector_type(8)))  float  v8f;
typedef __attribute__((ext_vector_type(4)))  int    i4v;

// CDNA5 async global->LDS path (ASYNCcnt-tracked DMA), gated on builtin availability.
#if defined(__HIP_DEVICE_COMPILE__) && defined(__gfx1250__) && \
    __has_builtin(__builtin_amdgcn_global_load_async_to_lds_b128)
#define HAVE_ASYNC 1
#else
#define HAVE_ASYNC 0
#endif

__device__ __forceinline__ int lane_id() { return threadIdx.x & 31; }

__device__ __forceinline__ v8f wmma_bf16(v16bf a, v16bf b, v8f c) {
  return __builtin_amdgcn_wmma_f32_16x16x32_bf16(false, a, false, b, (short)0, c, false, false);
}

#if HAVE_ASYNC
__device__ __forceinline__ void async_b128(const void* g, void* l) {
  __builtin_amdgcn_global_load_async_to_lds_b128(
      (__attribute__((address_space(1))) i4v*)(void*)g,
      (__attribute__((address_space(3))) i4v*)l, 0, 0);
}
__device__ __forceinline__ void wait_async() {
#if __has_builtin(__builtin_amdgcn_s_wait_asynccnt)
  __builtin_amdgcn_s_wait_asynccnt(0);
#else
  asm volatile("s_wait_asynccnt 0" ::: "memory");
#endif
}
#endif

// A-fragment (16x32 bf16) gather from LDS tile laid out [rows][32] row-major.
// Lane l holds row mbase+(l&15); lanes<16 hold K {0..7,16..23}, lanes>=16 hold K {8..15,24..31}.
__device__ __forceinline__ v16bf ldsA(const __bf16* As, int mbase) {
  int l = lane_id();
  int m = mbase + (l & 15);
  int khalf = (l & 16) ? 8 : 0;
  const __bf16* p = As + m * 32;
  v16bf a;
#pragma unroll
  for (int i = 0; i < 8; i++) { a[i] = p[khalf + i]; a[8 + i] = p[16 + khalf + i]; }
  return a;
}

// B-fragment (32x16 bf16) gather from LDS tile laid out [cols][32] (each col's K contiguous).
// Lanes<16 hold K {0..7,16..23}, lanes>=16 hold K {8..15,24..31}.
__device__ __forceinline__ v16bf ldsB(const __bf16* Bs, int nbase) {
  int l = lane_id();
  int n = nbase + (l & 15);
  int koff = (l & 16) ? 8 : 0;
  const __bf16* p = Bs + n * 32 + koff;
  v16bf b;
#pragma unroll
  for (int i = 0; i < 8; i++) { b[i] = p[i]; b[8 + i] = p[16 + i]; }
  return b;
}

// Load all 8 B-fragments into registers, then issue the WMMA chain (pipelines ds_load vs wmma).
__device__ __forceinline__ void mma_step8(const __bf16* sA, const __bf16* sB, int wave, v8f acc[8]) {
  v16bf a = ldsA(sA, wave * 16);
  v16bf b[8];
#pragma unroll
  for (int cf = 0; cf < 8; cf++) b[cf] = ldsB(sB, cf * 16);
#pragma unroll
  for (int cf = 0; cf < 8; cf++) acc[cf] = wmma_bf16(a, b[cf], acc[cf]);
}

// ---------------- cast fp32 -> bf16 ----------------
__global__ void cast_bf16_kernel(const float* __restrict__ src, __bf16* __restrict__ dst, size_t n) {
  size_t i = (size_t)blockIdx.x * 256 + threadIdx.x;
  if (i < n) dst[i] = (__bf16)src[i];
}

// ---------------- generic projection GEMM: C = A(bf16,[M,K]) * Bw(bf16,[N,K])^T + bias --------
// Double-buffered LDS staging; async DMA path on CDNA5 when available.
// mode 1: scatter in-proj output into z / xBC_raw / dt_raw.  mode 2: plain store to o0[M,N].
__global__ __launch_bounds__(256) void gemm_bf16_kernel(
    const __bf16* __restrict__ A, const __bf16* __restrict__ Bw,
    int N, int K, const float* __restrict__ bias, int mode,
    float* __restrict__ o0, float* __restrict__ o1, float* __restrict__ o2) {
  __shared__ __bf16 As[2][128 * 32];
  __shared__ __bf16 Bs[2][128 * 32];
  int m0 = blockIdx.y * 128;
  int n0 = blockIdx.x * 128;
  int t = threadIdx.x;
  int wave = t >> 5;
  int lane = t & 31;
  int sr = t >> 1, sh = (t & 1) * 16;   // staging row / 16-element half

  v8f acc[8];
#pragma unroll
  for (int i = 0; i < 8; i++)
#pragma unroll
    for (int r = 0; r < 8; r++) acc[i][r] = 0.f;

#if HAVE_ASYNC
  {   // prologue: DMA tile 0 straight into LDS
    const __bf16* ga = A + (size_t)(m0 + sr) * K + sh;
    __bf16* sa = As[0] + sr * 32 + sh;
    async_b128(ga, sa); async_b128(ga + 8, sa + 8);
    __bf16* sb = Bs[0] + sr * 32 + sh;
    if (n0 + sr < N) {
      const __bf16* gb = Bw + (size_t)(n0 + sr) * K + sh;
      async_b128(gb, sb); async_b128(gb + 8, sb + 8);
    } else {
      uint4 zz = {0, 0, 0, 0};
      ((uint4*)sb)[0] = zz; ((uint4*)sb)[1] = zz;
    }
  }
  wait_async();
  __syncthreads();
  for (int k0 = 0; k0 < K; k0 += 32) {
    int cur = (k0 >> 5) & 1;
    if (k0 + 32 < K) {   // DMA next tile into the other buffer while we compute
      const __bf16* ga = A + (size_t)(m0 + sr) * K + k0 + 32 + sh;
      __bf16* sa = As[cur ^ 1] + sr * 32 + sh;
      async_b128(ga, sa); async_b128(ga + 8, sa + 8);
      __bf16* sb = Bs[cur ^ 1] + sr * 32 + sh;
      if (n0 + sr < N) {
        const __bf16* gb = Bw + (size_t)(n0 + sr) * K + k0 + 32 + sh;
        async_b128(gb, sb); async_b128(gb + 8, sb + 8);
      } else {
        uint4 zz = {0, 0, 0, 0};
        ((uint4*)sb)[0] = zz; ((uint4*)sb)[1] = zz;
      }
    }
    mma_step8(As[cur], Bs[cur], wave, acc);
    wait_async();
    __syncthreads();
  }
#else
  {   // prologue: stage tile 0
    const uint4* g4 = (const uint4*)(A + (size_t)(m0 + sr) * K + sh);
    uint4* s4 = (uint4*)(As[0] + sr * 32 + sh);
    s4[0] = g4[0]; s4[1] = g4[1];
    uint4 v0 = {0, 0, 0, 0}, v1 = {0, 0, 0, 0};
    if (n0 + sr < N) {
      const uint4* h4 = (const uint4*)(Bw + (size_t)(n0 + sr) * K + sh);
      v0 = h4[0]; v1 = h4[1];
    }
    uint4* sb4 = (uint4*)(Bs[0] + sr * 32 + sh);
    sb4[0] = v0; sb4[1] = v1;
  }
  __syncthreads();
  for (int k0 = 0; k0 < K; k0 += 32) {
    int cur = (k0 >> 5) & 1;
    bool nxt = (k0 + 32 < K);
    uint4 ra0, ra1, rb0, rb1;
    if (nxt) {   // issue next-tile global loads before the WMMA block
      const uint4* g4 = (const uint4*)(A + (size_t)(m0 + sr) * K + k0 + 32 + sh);
      ra0 = g4[0]; ra1 = g4[1];
      uint4 zz = {0, 0, 0, 0};
      rb0 = zz; rb1 = zz;
      if (n0 + sr < N) {
        const uint4* h4 = (const uint4*)(Bw + (size_t)(n0 + sr) * K + k0 + 32 + sh);
        rb0 = h4[0]; rb1 = h4[1];
      }
    }
    mma_step8(As[cur], Bs[cur], wave, acc);
    if (nxt) {
      uint4* sa4 = (uint4*)(As[cur ^ 1] + sr * 32 + sh);
      sa4[0] = ra0; sa4[1] = ra1;
      uint4* sb4 = (uint4*)(Bs[cur ^ 1] + sr * 32 + sh);
      sb4[0] = rb0; sb4[1] = rb1;
    }
    __syncthreads();
  }
#endif

  int rlo = (lane >> 4) * 8;
  int col0 = n0 + (lane & 15);
#pragma unroll
  for (int cf = 0; cf < 8; cf++) {
    int col = col0 + cf * 16;
    if (col >= N) continue;
    float bv = bias[col];
#pragma unroll
    for (int r = 0; r < 8; r++) {
      int row = m0 + wave * 16 + rlo + r;
      float v = acc[cf][r] + bv;
      if (mode == 1) {
        if (col < DSSM)               o0[(size_t)row * DSSM + col] = v;
        else if (col < DSSM + CONVD)  o1[(size_t)row * CONVD + (col - DSSM)] = v;
        else                          o2[(size_t)row * NH + (col - DSSM - CONVD)] = v;
      } else {
        o0[(size_t)row * N + col] = v;
      }
    }
  }
}

// ---------------- depthwise causal conv (K=4) + SiLU ----------------
__global__ void conv_silu_kernel(const float* __restrict__ xr, const float* __restrict__ cw,
                                 const float* __restrict__ cb, float* __restrict__ xc) {
  size_t idx = (size_t)blockIdx.x * 256 + threadIdx.x;
  if (idx >= (size_t)BL * CONVD) return;
  int c = (int)(idx % CONVD);
  size_t row = idx / CONVD;
  int l = (int)(row % L_SZ);
  float acc = cb[c];
#pragma unroll
  for (int k = 0; k < KCONV; k++) {
    int ls = l - (KCONV - 1) + k;
    if (ls >= 0) acc += xr[(row - (KCONV - 1) + k) * CONVD + c] * cw[c * KCONV + k];
  }
  xc[idx] = acc / (1.f + __expf(-acc));
}

// ---------------- dt = softplus(dt_raw + bias), per-chunk cumsum of dt*A ----------------
__global__ void dt_cumsum_kernel(const float* __restrict__ dtraw, const float* __restrict__ dt_bias,
                                 const float* __restrict__ A_log, float* __restrict__ dt,
                                 float* __restrict__ dAcs) {
  int bc = blockIdx.x;
  int h = threadIdx.x;
  float Ah = -__expf(A_log[h]);
  float bia = dt_bias[h];
  size_t base = (size_t)bc * CHUNK;
  float cs = 0.f;
  for (int i = 0; i < CHUNK; i++) {
    size_t row = base + i;
    float x = dtraw[row * NH + h] + bia;
    float d = (x > 20.f) ? x : log1pf(__expf(x));
    dt[row * NH + h] = d;
    cs += d * Ah;
    dAcs[row * NH + h] = cs;
  }
}

// ---------------- G[bc][i][j] = sum_n C[i,n]*B[j,n] (256x256x128 per (b,chunk)) ----------------
__global__ __launch_bounds__(256) void g_kernel(const float* __restrict__ xc, float* __restrict__ G) {
  int bc = blockIdx.z;
  int i0 = blockIdx.y * 128;
  int j0 = blockIdx.x * 128;
  size_t rowbase = (size_t)bc * CHUNK;
  __shared__ __bf16 As[128 * 32];
  __shared__ __bf16 Bs[128 * 32];
  int t = threadIdx.x, wave = t >> 5, lane = t & 31;

  v8f acc[8];
#pragma unroll
  for (int i = 0; i < 8; i++)
#pragma unroll
    for (int r = 0; r < 8; r++) acc[i][r] = 0.f;

  for (int k0 = 0; k0 < DST; k0 += 32) {
    int r = t >> 1, h = (t & 1) * 16;
    const float* ga = xc + (rowbase + i0 + r) * CONVD + (DSSM + DST) + k0 + h;  // C block
    const float* gb = xc + (rowbase + j0 + r) * CONVD + DSSM + k0 + h;          // B block
#pragma unroll
    for (int i = 0; i < 16; i++) { As[r * 32 + h + i] = (__bf16)ga[i]; Bs[r * 32 + h + i] = (__bf16)gb[i]; }
    __syncthreads();
    mma_step8(As, Bs, wave, acc);
    __syncthreads();
  }

  float* Gb = G + (size_t)bc * CHUNK * CHUNK;
  int rlo = (lane >> 4) * 8, jc = j0 + (lane & 15);
#pragma unroll
  for (int cf = 0; cf < 8; cf++) {
    int j = jc + cf * 16;
#pragma unroll
    for (int r = 0; r < 8; r++) {
      int i = i0 + wave * 16 + rlo + r;
      Gb[(size_t)i * CHUNK + j] = acc[cf][r];
    }
  }
}

// ---------------- per-(b,chunk,head) states[p][n] = sum_l x[l,p]*w[l]*B[l,n] ----------------
__global__ __launch_bounds__(256) void states_kernel(const float* __restrict__ xc,
                                                     const float* __restrict__ dt,
                                                     const float* __restrict__ dAcs,
                                                     float* __restrict__ states) {
  int h = blockIdx.x;
  int bc = blockIdx.y;
  size_t rowbase = (size_t)bc * CHUNK;
  __shared__ float wls[CHUNK];
  __shared__ __bf16 As[64 * 32];    // [p][lk], decay*dt pre-applied
  __shared__ __bf16 Bs[128 * 32];   // [n][lk]
  int t = threadIdx.x, wave = t >> 5, lane = t & 31;
  int hcol = h * HD;

  {
    float alast = dAcs[(rowbase + CHUNK - 1) * NH + h];
    float a = dAcs[(rowbase + t) * NH + h];
    wls[t] = __expf(alast - a) * dt[(rowbase + t) * NH + h];
  }
  __syncthreads();

  v8f acc[4];
#pragma unroll
  for (int i = 0; i < 4; i++)
#pragma unroll
    for (int r = 0; r < 8; r++) acc[i][r] = 0.f;

  for (int k0 = 0; k0 < CHUNK; k0 += 32) {
    {
      int lk = t >> 3, pb = (t & 7) * 8;
      const float* g = xc + (rowbase + k0 + lk) * CONVD + hcol + pb;
      float wl = wls[k0 + lk];
#pragma unroll
      for (int i = 0; i < 8; i++) As[(pb + i) * 32 + lk] = (__bf16)(g[i] * wl);
    }
    {
      int lk = t >> 3, nb = (t & 7) * 16;
      const float* g = xc + (rowbase + k0 + lk) * CONVD + DSSM + nb;
#pragma unroll
      for (int i = 0; i < 16; i++) Bs[(nb + i) * 32 + lk] = (__bf16)g[i];
    }
    __syncthreads();
    v16bf a = ldsA(As, (wave & 3) * 16);
    int nbase = (wave >> 2) * 64;
    v16bf bb[4];
#pragma unroll
    for (int cf = 0; cf < 4; cf++) bb[cf] = ldsB(Bs, nbase + cf * 16);
#pragma unroll
    for (int cf = 0; cf < 4; cf++) acc[cf] = wmma_bf16(a, bb[cf], acc[cf]);
    __syncthreads();
  }

  float* out = states + ((size_t)bc * NH + h) * HD * DST;
  int rlo = (lane >> 4) * 8;
  int p0 = (wave & 3) * 16;
  int n0 = (wave >> 2) * 64 + (lane & 15);
#pragma unroll
  for (int cf = 0; cf < 4; cf++) {
    int n = n0 + cf * 16;
#pragma unroll
    for (int r = 0; r < 8; r++) out[(size_t)(p0 + rlo + r) * DST + n] = acc[cf][r];
  }
}

// ---------------- inter-chunk scan ----------------
__global__ void scan_kernel(const float* __restrict__ states, const float* __restrict__ dAcs,
                            float* __restrict__ prevst) {
  int bh = blockIdx.x;
  int b = bh / NH, h = bh % NH;
  int t = threadIdx.x;
  float carry[32];
#pragma unroll
  for (int i = 0; i < 32; i++) carry[i] = 0.f;
  for (int c = 0; c < NC; c++) {
    size_t off = ((size_t)(b * NC + c) * NH + h) * (HD * DST);
#pragma unroll
    for (int i = 0; i < 32; i++) prevst[off + t + i * 256] = carry[i];
    float dec = __expf(dAcs[((size_t)(b * NC + c) * CHUNK + CHUNK - 1) * NH + h]);
#pragma unroll
    for (int i = 0; i < 32; i++) carry[i] = dec * carry[i] + states[off + t + i * 256];
  }
}

// ---------------- fused Y = Y_diag + Y_off + D*x ----------------
__global__ __launch_bounds__(256) void y_kernel(const float* __restrict__ xc,
                                                const float* __restrict__ G,
                                                const float* __restrict__ dt,
                                                const float* __restrict__ dAcs,
                                                const float* __restrict__ prevst,
                                                const float* __restrict__ Dp,
                                                float* __restrict__ Y) {
  int ihalf = blockIdx.x;
  int h = blockIdx.y;
  int bc = blockIdx.z;
  size_t rowbase = (size_t)bc * CHUNK;
  int t = threadIdx.x, wave = t >> 5, lane = t & 31;
  __shared__ float aS[CHUNK], dtS[CHUNK];
  __shared__ __bf16 xT[64 * 32];
  aS[t] = dAcs[(rowbase + t) * NH + h];
  dtS[t] = dt[(rowbase + t) * NH + h];
  __syncthreads();

  int iloc = ihalf * 128 + wave * 16;
  int hcol = h * HD;
  const float* Gb = G + (size_t)bc * CHUNK * CHUNK;

  v8f acc[4];
#pragma unroll
  for (int i = 0; i < 4; i++)
#pragma unroll
    for (int r = 0; r < 8; r++) acc[i][r] = 0.f;

  // ---- intra-chunk (diag): K = j over 256 ----
  for (int k0 = 0; k0 < CHUNK; k0 += 32) {
    {
      int jk = t >> 3, pb = (t & 7) * 8;
      const float* g = xc + (rowbase + k0 + jk) * CONVD + hcol + pb;
#pragma unroll
      for (int i = 0; i < 8; i++) xT[(pb + i) * 32 + jk] = (__bf16)g[i];
    }
    __syncthreads();
    int i = iloc + (lane & 15);
    int khalf = (lane & 16) ? 8 : 0;
    float ai = aS[i];
    const float* gr = Gb + (size_t)i * CHUNK + k0;
    v16bf a;
#pragma unroll
    for (int e = 0; e < 16; e++) {
      int jk = (e < 8) ? (khalf + e) : (16 + khalf + e - 8);
      int j = k0 + jk;
      float v = 0.f;
      if (j <= i) v = gr[jk] * __expf(ai - aS[j]) * dtS[j];   // exponent <= 0: stable
      a[e] = (__bf16)v;
    }
    v16bf bb[4];
#pragma unroll
    for (int cf = 0; cf < 4; cf++) bb[cf] = ldsB(xT, cf * 16);
#pragma unroll
    for (int cf = 0; cf < 4; cf++) acc[cf] = wmma_bf16(a, bb[cf], acc[cf]);
    __syncthreads();
  }

  // ---- inter-chunk (off): K = n over 128, B = prev_states[p][n] ----
  {
    int i = iloc + (lane & 15);
    float eai = __expf(aS[i]);
    const float* pv = prevst + ((size_t)bc * NH + h) * (HD * DST);
    int khalf = (lane & 16) ? 8 : 0;
    int koff = (lane & 16) ? 8 : 0;
    for (int k0 = 0; k0 < DST; k0 += 32) {
      const float* cr = xc + (rowbase + i) * CONVD + (DSSM + DST) + k0;
      v16bf a;
#pragma unroll
      for (int e = 0; e < 16; e++) {
        int nk = (e < 8) ? (khalf + e) : (16 + khalf + e - 8);
        a[e] = (__bf16)(cr[nk] * eai);
      }
      v16bf bfr[4];
#pragma unroll
      for (int cf = 0; cf < 4; cf++) {
        int p = cf * 16 + (lane & 15);
        const float* pr = pv + (size_t)p * DST + k0 + koff;
#pragma unroll
        for (int e = 0; e < 16; e++) bfr[cf][e] = (__bf16)pr[(e < 8) ? e : (8 + e)];
      }
#pragma unroll
      for (int cf = 0; cf < 4; cf++) acc[cf] = wmma_bf16(a, bfr[cf], acc[cf]);
    }
  }

  float Dh = Dp[h];
  int rlo = (lane >> 4) * 8;
  int pl = lane & 15;
#pragma unroll
  for (int cf = 0; cf < 4; cf++) {
    int p = cf * 16 + pl;
#pragma unroll
    for (int r = 0; r < 8; r++) {
      size_t row = rowbase + iloc + rlo + r;
      float xv = xc[row * CONVD + hcol + p];
      Y[row * DSSM + hcol + p] = acc[cf][r] + Dh * xv;
    }
  }
}

// ---------------- gating + RMSNorm, cast to bf16 for out-proj ----------------
__global__ __launch_bounds__(256) void gate_norm_kernel(const float* __restrict__ Y,
                                                        const float* __restrict__ z,
                                                        const float* __restrict__ nw,
                                                        __bf16* __restrict__ yB) {
  int row = blockIdx.x;
  int t = threadIdx.x;
  __shared__ float red[256];
  float vals[8];
  float ss = 0.f;
#pragma unroll
  for (int i = 0; i < 8; i++) {
    int c = t + i * 256;
    float zv = z[(size_t)row * DSSM + c];
    float v = Y[(size_t)row * DSSM + c] * (zv / (1.f + __expf(-zv)));
    vals[i] = v;
    ss += v * v;
  }
  red[t] = ss;
  __syncthreads();
  for (int s = 128; s > 0; s >>= 1) { if (t < s) red[t] += red[t + s]; __syncthreads(); }
  float scale = rsqrtf(red[0] / (float)DSSM + 1e-12f);
#pragma unroll
  for (int i = 0; i < 8; i++) {
    int c = t + i * 256;
    yB[(size_t)row * DSSM + c] = (__bf16)(vals[i] * scale * nw[c]);
  }
}

// ---------------- host launcher ----------------
extern "C" void kernel_launch(void* const* d_in, const int* in_sizes, int n_in,
                              void* d_out, int out_size, void* d_ws, size_t ws_size,
                              hipStream_t stream) {
  const float* u      = (const float*)d_in[0];
  const float* W_in   = (const float*)d_in[1];
  const float* b_in   = (const float*)d_in[2];
  const float* conv_w = (const float*)d_in[3];
  const float* conv_b = (const float*)d_in[4];
  const float* dtb    = (const float*)d_in[5];
  const float* A_log  = (const float*)d_in[6];
  const float* Dp     = (const float*)d_in[7];
  const float* norm_w = (const float*)d_in[8];
  const float* W_out  = (const float*)d_in[9];
  const float* b_out  = (const float*)d_in[10];
  float* out = (float*)d_out;

  char* p = (char*)d_ws;
  auto alloc = [&](size_t bytes) -> void* {
    void* r = (void*)p;
    p += (bytes + 255) & ~(size_t)255;
    return r;
  };
  __bf16* uB    = (__bf16*)alloc((size_t)BL * DM * 2);
  __bf16* WinB  = (__bf16*)alloc((size_t)DINP * DM * 2);
  __bf16* WoutB = (__bf16*)alloc((size_t)DM * DIN * 2);
  float*  z     = (float*)alloc((size_t)BL * DSSM * 4);
  float*  xBCr  = (float*)alloc((size_t)BL * CONVD * 4);
  float*  dtraw = (float*)alloc((size_t)BL * NH * 4);
  float*  xBCc  = (float*)alloc((size_t)BL * CONVD * 4);
  float*  dt    = (float*)alloc((size_t)BL * NH * 4);
  float*  dAcs  = (float*)alloc((size_t)BL * NH * 4);
  float*  G     = (float*)alloc((size_t)BC_N * CHUNK * CHUNK * 4);
  float*  st    = (float*)alloc((size_t)BC_N * NH * HD * DST * 4);
  float*  pst   = (float*)alloc((size_t)BC_N * NH * HD * DST * 4);
  float*  Y     = (float*)alloc((size_t)BL * DSSM * 4);
  __bf16* yB    = (__bf16*)alloc((size_t)BL * DSSM * 2);

  {
    size_t n = (size_t)BL * DM;
    cast_bf16_kernel<<<(unsigned)((n + 255) / 256), 256, 0, stream>>>(u, uB, n);
    n = (size_t)DINP * DM;
    cast_bf16_kernel<<<(unsigned)((n + 255) / 256), 256, 0, stream>>>(W_in, WinB, n);
    n = (size_t)DM * DIN;
    cast_bf16_kernel<<<(unsigned)((n + 255) / 256), 256, 0, stream>>>(W_out, WoutB, n);
  }

  gemm_bf16_kernel<<<dim3((DINP + 127) / 128, BL / 128), 256, 0, stream>>>(
      uB, WinB, DINP, DM, b_in, 1, z, xBCr, dtraw);

  {
    size_t n = (size_t)BL * CONVD;
    conv_silu_kernel<<<(unsigned)((n + 255) / 256), 256, 0, stream>>>(xBCr, conv_w, conv_b, xBCc);
  }

  dt_cumsum_kernel<<<BC_N, NH, 0, stream>>>(dtraw, dtb, A_log, dt, dAcs);

  g_kernel<<<dim3(2, 2, BC_N), 256, 0, stream>>>(xBCc, G);

  states_kernel<<<dim3(NH, BC_N), 256, 0, stream>>>(xBCc, dt, dAcs, st);

  scan_kernel<<<B_SZ * NH, 256, 0, stream>>>(st, dAcs, pst);

  y_kernel<<<dim3(2, NH, BC_N), 256, 0, stream>>>(xBCc, G, dt, dAcs, pst, Dp, Y);

  gate_norm_kernel<<<BL, 256, 0, stream>>>(Y, z, norm_w, yB);

  gemm_bf16_kernel<<<dim3(DM / 128, BL / 128), 256, 0, stream>>>(
      yB, WoutB, DM, DIN, b_out, 2, out, nullptr, nullptr);
}
